// SNLIClassifier_29497835389440
// MI455X (gfx1250) — compile-verified, matrix operand order checked by reading, not run on hardware
//
#include <hip/hip_runtime.h>
#include <hip/hip_bf16.h>
#include <math.h>

// ---------------------------------------------------------------------------
// Types for CDNA5 WMMA (gfx1250, wave32)
// ---------------------------------------------------------------------------
typedef __attribute__((ext_vector_type(16))) _Float16 v16h;
typedef __attribute__((ext_vector_type(8)))  _Float16 v8h;
typedef __attribute__((ext_vector_type(8)))  float    v8f;

#define B_   128
#define T_   128
#define E_   512
#define HH_  512            // Hh = HID/2
#define G_   2048           // 4*Hh gate width
#define KC_  1024           // E + Hh (concat K for fused gates GEMM)
#define HDIM_ 1024
#define FDIM_ 4096          // 4*HID feature width

__device__ __forceinline__ float sigmoidf_(float x) { return 1.0f / (1.0f + __expf(-x)); }

// ---------------------------------------------------------------------------
// WMMA fragment loader.
// 16-bit A (16x32 MxK) layout per CDNA5 ISA 7.12.2: lane = half*16 + m,
//   element j<8  -> K = kBase + 8*half + j
//   element j>=8 -> K = kBase + 16 + 8*half + (j-8)
// i.e. two contiguous 16-byte chunks per lane.  B (32x16 KxN) symmetric with
// n = lane%16 reading rows of the [N,K] weight matrix.
// ---------------------------------------------------------------------------
__device__ __forceinline__ v16h load_frag_row(const _Float16* __restrict__ row,
                                              int kBase, int half) {
    v8h lo = *(const v8h*)(row + kBase + 8 * half);
    v8h hi = *(const v8h*)(row + kBase + 16 + 8 * half);
    v16h r;
#pragma unroll
    for (int j = 0; j < 8; ++j) { r[j] = lo[j]; r[8 + j] = hi[j]; }
    return r;
}

// ---------------------------------------------------------------------------
// Generic GEMM (compile-time K):  C[M,HDIM] = A[M,K](f16) * W[HDIM,K]^T (+b)(+relu)
// One wave -> 64(M) x 64(N) tile: acc[4][4] (128 VGPRs).  __launch_bounds__
// (128,1) lifts the compiler's 128-VGPR occupancy cap so the tile lives
// entirely in registers (no scratch spills).  16 WMMA per 16 b128 loads.
// block = 128 (4 waves, distinct N subtiles); grid = (N/256, M/64).
// ---------------------------------------------------------------------------
template<int K>
__global__ __launch_bounds__(128, 1)
void wmma_gemm_f16(const _Float16* __restrict__ A,
                   const _Float16* __restrict__ W,
                   float* __restrict__ C,
                   const float* __restrict__ bias, int relu) {
    const int lane  = threadIdx.x & 31;
    const int wave  = threadIdx.x >> 5;
    const int half  = lane >> 4;
    const int l16   = lane & 15;
    const int nTile = (blockIdx.x * 4 + wave) * 64;
    const int mTile = blockIdx.y * 64;

    const _Float16* aB = A + (size_t)(mTile + l16) * K;
    const _Float16* wB = W + (size_t)(nTile + l16) * K;

    v8f acc[4][4];
#pragma unroll
    for (int mi = 0; mi < 4; ++mi)
#pragma unroll
        for (int ni = 0; ni < 4; ++ni) acc[mi][ni] = v8f{};

    for (int k0 = 0; k0 < K; k0 += 32) {
        v16h a[4];
#pragma unroll
        for (int mi = 0; mi < 4; ++mi)
            a[mi] = load_frag_row(aB + mi * 16 * K, k0, half);
#pragma unroll
        for (int ni = 0; ni < 4; ++ni) {
            v16h b = load_frag_row(wB + ni * 16 * K, k0, half);
#pragma unroll
            for (int mi = 0; mi < 4; ++mi)
                acc[mi][ni] = __builtin_amdgcn_wmma_f32_16x16x32_f16(
                    false, a[mi], false, b, (short)0, acc[mi][ni], false, false);
        }
    }
#pragma unroll
    for (int ni = 0; ni < 4; ++ni) {
        int n = nTile + ni * 16 + l16;
        float bv = bias ? bias[n] : 0.0f;
#pragma unroll
        for (int mi = 0; mi < 4; ++mi) {
#pragma unroll
            for (int r = 0; r < 8; ++r) {
                int m = mTile + mi * 16 + r + 8 * half;
                float v = acc[mi][ni][r] + bv;
                if (relu) v = fmaxf(v, 0.0f);
                C[(size_t)m * HDIM_ + n] = v;
            }
        }
    }
}

// ---------------------------------------------------------------------------
// Fused LSTM gates GEMM for one timestep (the serial critical path):
//   rows: run r = row/128  (0:hyp-fwd 1:hyp-bwd 2:prem-fwd 3:prem-bwd), b = row%128
//   A[row, k] = k<512 ? xin[sent][tt][b][k] : H[row][k-512]   (tt = fwd? t : T-1-t)
//   gates[512,2048] = A @ Wcat[2048,1024]^T      (Wcat = [W_ih | W_hh] f16)
// 64x64 per wave; K-loop split into x-phase [0,512) and h-phase [512,1024).
// Single base pointer per operand; grid = (8,8), block = 128 (256 waves).
// __launch_bounds__(128,1): full VGPR budget, no spills on the serial path.
// ---------------------------------------------------------------------------
__global__ __launch_bounds__(128, 1)
void lstm_gates_gemm(const _Float16* __restrict__ xin,   // [2][T][B][E] f16
                     const _Float16* __restrict__ H,     // [512][HH] f16
                     const _Float16* __restrict__ Wcat,  // [2048][1024] f16
                     float* __restrict__ gates,          // [512][2048] f32
                     int t) {
    const int lane  = threadIdx.x & 31;
    const int wave  = threadIdx.x >> 5;
    const int half  = lane >> 4;
    const int l16   = lane & 15;
    const int nTile = (blockIdx.x * 4 + wave) * 64;
    const int mTile = blockIdx.y * 64;

    const int run  = mTile >> 7;               // 64-tile sits inside one run
    const int fwd  = ((run & 1) == 0);
    const int sent = run >> 1;
    const int tt   = fwd ? t : (T_ - 1 - t);

    const _Float16* xB = xin + (((size_t)sent * T_ + tt) * B_ + ((mTile & 127) + l16)) * E_;
    const _Float16* hB = H + (size_t)(mTile + l16) * HH_;
    const _Float16* wB = Wcat + (size_t)(nTile + l16) * KC_;

    v8f acc[4][4];
#pragma unroll
    for (int mi = 0; mi < 4; ++mi)
#pragma unroll
        for (int ni = 0; ni < 4; ++ni) acc[mi][ni] = v8f{};

    // phase 1: k in [0, 512) from x_t
    for (int k0 = 0; k0 < E_; k0 += 32) {
        v16h a[4];
#pragma unroll
        for (int mi = 0; mi < 4; ++mi)
            a[mi] = load_frag_row(xB + mi * 16 * E_, k0, half);
#pragma unroll
        for (int ni = 0; ni < 4; ++ni) {
            v16h b = load_frag_row(wB + ni * 16 * KC_, k0, half);
#pragma unroll
            for (int mi = 0; mi < 4; ++mi)
                acc[mi][ni] = __builtin_amdgcn_wmma_f32_16x16x32_f16(
                    false, a[mi], false, b, (short)0, acc[mi][ni], false, false);
        }
    }
    // phase 2: k in [512, 1024) from h_{t-1}
    for (int k0 = 0; k0 < HH_; k0 += 32) {
        v16h a[4];
#pragma unroll
        for (int mi = 0; mi < 4; ++mi)
            a[mi] = load_frag_row(hB + mi * 16 * HH_, k0, half);
#pragma unroll
        for (int ni = 0; ni < 4; ++ni) {
            v16h b = load_frag_row(wB + ni * 16 * KC_, E_ + k0, half);
#pragma unroll
            for (int mi = 0; mi < 4; ++mi)
                acc[mi][ni] = __builtin_amdgcn_wmma_f32_16x16x32_f16(
                    false, a[mi], false, b, (short)0, acc[mi][ni], false, false);
        }
    }
#pragma unroll
    for (int ni = 0; ni < 4; ++ni) {
        int n = nTile + ni * 16 + l16;
#pragma unroll
        for (int mi = 0; mi < 4; ++mi) {
#pragma unroll
            for (int r = 0; r < 8; ++r) {
                int m = mTile + mi * 16 + r + 8 * half;
                gates[(size_t)m * G_ + n] = acc[mi][ni][r];
            }
        }
    }
}

// LSTM cell pointwise: i,f,g,o chunks of 512; updates H(f16) and Cst(f32) in place.
__global__ void lstm_cell(const float* __restrict__ gates,
                          const float* __restrict__ b_ih, const float* __restrict__ b_hh,
                          _Float16* __restrict__ H, float* __restrict__ Cst) {
    int idx = blockIdx.x * blockDim.x + threadIdx.x;      // 512*512
    if (idx >= 512 * HH_) return;
    int row = idx >> 9, j = idx & 511;
    const float* gr = gates + (size_t)row * G_;
    float gi = gr[j]          + b_ih[j]          + b_hh[j];
    float gf = gr[512 + j]    + b_ih[512 + j]    + b_hh[512 + j];
    float gg = gr[1024 + j]   + b_ih[1024 + j]   + b_hh[1024 + j];
    float go = gr[1536 + j]   + b_ih[1536 + j]   + b_hh[1536 + j];
    float c  = sigmoidf_(gf) * Cst[idx] + sigmoidf_(gi) * tanhf(gg);
    float h  = sigmoidf_(go) * tanhf(c);
    Cst[idx] = c;
    H[idx]   = (_Float16)h;
}

// Embedding gather -> time-major f16 activations xin[2][T][B][E]
__global__ void gather_embed(const int* __restrict__ hyp, const int* __restrict__ prem,
                             const float* __restrict__ embed, _Float16* __restrict__ xin) {
    size_t idx = (size_t)blockIdx.x * blockDim.x + threadIdx.x;  // 2*T*B*E
    int e = idx & 511, b = (idx >> 9) & 127, t = (idx >> 16) & 127, s = (int)(idx >> 23);
    int tok = (s == 0 ? hyp : prem)[b * T_ + t];
    xin[idx] = (_Float16)embed[(size_t)tok * E_ + e];
}

// Pack [W_ih | W_hh] (fp32) -> Wcat f16 [2048][1024]
__global__ void pack_wcat(const float* __restrict__ Wih, const float* __restrict__ Whh,
                          _Float16* __restrict__ Wcat) {
    int idx = blockIdx.x * blockDim.x + threadIdx.x;              // 2048*1024
    if (idx >= G_ * KC_) return;
    int n = idx >> 10, k = idx & 1023;
    float v = (k < 512) ? Wih[n * 512 + k] : Whh[n * 512 + (k - 512)];
    Wcat[idx] = (_Float16)v;
}

__global__ void pack_f16(const float* __restrict__ src, _Float16* __restrict__ dst, int n) {
    int idx = blockIdx.x * blockDim.x + threadIdx.x;
    if (idx < n) dst[idx] = (_Float16)src[idx];
}

__global__ void zero_b32(uint32_t* __restrict__ p, int n) {
    int idx = blockIdx.x * blockDim.x + threadIdx.x;
    if (idx < n) p[idx] = 0u;
}

// features f[128,4096] = [hyp, prem, prem-hyp, hyp*prem] from final states H
__global__ void build_features(const _Float16* __restrict__ H, float* __restrict__ feat) {
    int idx = blockIdx.x * blockDim.x + threadIdx.x;              // 128*4096
    if (idx >= B_ * FDIM_) return;
    int b = idx >> 12, j = idx & 4095;
    int seg = j >> 10, u = j & 1023;
    float hy = (float)(u < 512 ? H[(size_t)(0 * B_ + b) * HH_ + u]
                               : H[(size_t)(1 * B_ + b) * HH_ + (u - 512)]);
    float pr = (float)(u < 512 ? H[(size_t)(2 * B_ + b) * HH_ + u]
                               : H[(size_t)(3 * B_ + b) * HH_ + (u - 512)]);
    float v = (seg == 0) ? hy : (seg == 1) ? pr : (seg == 2) ? (pr - hy) : hy * pr;
    feat[idx] = v;
}

// Row LayerNorm (population variance), fp32 in -> f16 out (feeds next WMMA GEMM)
__global__ void ln_rows(const float* __restrict__ x, const float* __restrict__ g,
                        const float* __restrict__ bt, _Float16* __restrict__ out, int width) {
    int row = blockIdx.x;
    const float* xr = x + (size_t)row * width;
    float s = 0.f, s2 = 0.f;
    for (int j = threadIdx.x; j < width; j += blockDim.x) {
        float v = xr[j]; s += v; s2 += v * v;
    }
    __shared__ float sh[512];
    sh[threadIdx.x] = s; sh[256 + threadIdx.x] = s2;
    __syncthreads();
    for (int st = 128; st > 0; st >>= 1) {
        if ((int)threadIdx.x < st) {
            sh[threadIdx.x] += sh[threadIdx.x + st];
            sh[256 + threadIdx.x] += sh[256 + threadIdx.x + st];
        }
        __syncthreads();
    }
    float mean = sh[0] / width;
    float var  = sh[256] / width - mean * mean;
    float rstd = rsqrtf(var + 1e-6f);
    _Float16* orow = out + (size_t)row * width;
    for (int j = threadIdx.x; j < width; j += blockDim.x)
        orow[j] = (_Float16)((xr[j] - mean) * rstd * g[j] + bt[j]);
}

// Tiny output head: logits[128,3] = a2 @ out_W^T + out_b
__global__ void out_proj(const _Float16* __restrict__ a2, const float* __restrict__ W,
                         const float* __restrict__ bias, float* __restrict__ out) {
    int idx = blockIdx.x * blockDim.x + threadIdx.x;
    if (idx >= B_ * 3) return;
    int b = idx / 3, c = idx % 3;
    const _Float16* ar = a2 + (size_t)b * HDIM_;
    const float* wr = W + (size_t)c * HDIM_;
    float s = bias[c];
    for (int j = 0; j < HDIM_; ++j) s += (float)ar[j] * wr[j];
    out[idx] = s;
}

// ---------------------------------------------------------------------------
extern "C" void kernel_launch(void* const* d_in, const int* in_sizes, int n_in,
                              void* d_out, int out_size, void* d_ws, size_t ws_size,
                              hipStream_t stream) {
    const int*   hyp_tok = (const int*)d_in[0];
    const int*   prm_tok = (const int*)d_in[1];
    const float* embed   = (const float*)d_in[2];
    const float* W_ih    = (const float*)d_in[3];
    const float* W_hh    = (const float*)d_in[4];
    const float* b_ih    = (const float*)d_in[5];
    const float* b_hh    = (const float*)d_in[6];
    const float* mlp1_W  = (const float*)d_in[7];
    const float* mlp1_b  = (const float*)d_in[8];
    const float* mlp2_W  = (const float*)d_in[9];
    const float* mlp2_b  = (const float*)d_in[10];
    const float* out_W   = (const float*)d_in[11];
    const float* out_b   = (const float*)d_in[12];
    const float* ln0_g = (const float*)d_in[13], *ln0_b = (const float*)d_in[14];
    const float* ln1_g = (const float*)d_in[15], *ln1_b = (const float*)d_in[16];
    const float* ln2_g = (const float*)d_in[17], *ln2_b = (const float*)d_in[18];

    // workspace layout (256B aligned)
    char* ws = (char*)d_ws; size_t off = 0;
    auto alloc = [&](size_t bytes) { char* p = ws + off; off += (bytes + 255) & ~(size_t)255; return p; };
    _Float16* xin   = (_Float16*)alloc((size_t)2 * T_ * B_ * E_ * 2);   // 32 MB
    _Float16* Wcat  = (_Float16*)alloc((size_t)G_ * KC_ * 2);           // 4 MB
    _Float16* W1h   = (_Float16*)alloc((size_t)HDIM_ * FDIM_ * 2);      // 8 MB
    _Float16* W2h   = (_Float16*)alloc((size_t)HDIM_ * HDIM_ * 2);      // 2 MB
    _Float16* H     = (_Float16*)alloc((size_t)512 * HH_ * 2);
    float*    Cst   = (float*)   alloc((size_t)512 * HH_ * 4);
    float*    gates = (float*)   alloc((size_t)512 * G_ * 4);
    float*    feat  = (float*)   alloc((size_t)B_ * FDIM_ * 4);
    _Float16* fA    = (_Float16*)alloc((size_t)B_ * FDIM_ * 2);
    float*    m1o   = (float*)   alloc((size_t)B_ * HDIM_ * 4);
    _Float16* a1    = (_Float16*)alloc((size_t)B_ * HDIM_ * 2);
    float*    m2o   = (float*)   alloc((size_t)B_ * HDIM_ * 4);
    _Float16* a2    = (_Float16*)alloc((size_t)B_ * HDIM_ * 2);
    if (off > ws_size) return;  // deterministic guard

    // 1) pack weights to f16
    pack_wcat<<<(G_ * KC_ + 255) / 256, 256, 0, stream>>>(W_ih, W_hh, Wcat);
    pack_f16 <<<(HDIM_ * FDIM_ + 255) / 256, 256, 0, stream>>>(mlp1_W, W1h, HDIM_ * FDIM_);
    pack_f16 <<<(HDIM_ * HDIM_ + 255) / 256, 256, 0, stream>>>(mlp2_W, W2h, HDIM_ * HDIM_);

    // 2) embedding gather (both sentences, time-major, f16)
    gather_embed<<<(2 * T_ * B_ * E_) / 256, 256, 0, stream>>>(hyp_tok, prm_tok, embed, xin);

    // 3) zero LSTM state (ws is poisoned)
    zero_b32<<<(512 * HH_ / 2 + 255) / 256, 256, 0, stream>>>((uint32_t*)H,  512 * HH_ / 2);
    zero_b32<<<(512 * HH_ + 255) / 256, 256, 0, stream>>>((uint32_t*)Cst, 512 * HH_);

    // 4) recurrence: 128 steps, 2 kernels each.  All 4 runs batched (M=512).
    for (int t = 0; t < T_; ++t) {
        lstm_gates_gemm<<<dim3(G_ / 256, 512 / 64), 128, 0, stream>>>(xin, H, Wcat, gates, t);
        lstm_cell<<<(512 * HH_) / 256, 256, 0, stream>>>(gates, b_ih, b_hh, H, Cst);
    }

    // 5) features -> ln0 -> mlp1(+relu) -> ln1 -> mlp2(+relu) -> ln2 -> head
    build_features<<<(B_ * FDIM_) / 256, 256, 0, stream>>>(H, feat);
    ln_rows<<<B_, 256, 0, stream>>>(feat, ln0_g, ln0_b, fA, FDIM_);
    wmma_gemm_f16<FDIM_><<<dim3(HDIM_ / 256, B_ / 64), 128, 0, stream>>>(
        fA, W1h, m1o, mlp1_b, 1);
    ln_rows<<<B_, 256, 0, stream>>>(m1o, ln1_g, ln1_b, a1, HDIM_);
    wmma_gemm_f16<HDIM_><<<dim3(HDIM_ / 256, B_ / 64), 128, 0, stream>>>(
        a1, W2h, m2o, mlp2_b, 1);
    ln_rows<<<B_, 256, 0, stream>>>(m2o, ln2_g, ln2_b, a2, HDIM_);
    out_proj<<<3, 128, 0, stream>>>(a2, out_W, out_b, (float*)d_out);
}
